// EdgeGNN_37632503448200
// MI455X (gfx1250) — matrix operand order repaired and checked by reading, not compiled, not run on hardware
//
#include <hip/hip_runtime.h>
#include <hip/hip_bf16.h>

typedef __attribute__((ext_vector_type(16))) _Float16 v16h;
typedef __attribute__((ext_vector_type(8)))  float    v8f;

namespace {
constexpr int  B_ = 2;
constexpr int  N_ = 100000;
constexpr int  F_ = 8;
constexpr int  E_ = 3200000;
constexpr int  TILES_PER_BATCH = E_ / 16;          // 200000 (E divisible by 16)
constexpr int  NTILES          = B_ * TILES_PER_BATCH;
constexpr int  WAVES_PER_BLOCK = 8;                // 256 threads = 8 wave32
constexpr int  EDGE_BLOCKS     = 4096;
}

__device__ __forceinline__ float fast_sigmoid(float x) {
  return 1.0f / (1.0f + __expf(-x));
}

// ---------------------------------------------------------------------------
// Kernel 1: zero the [B, N, 16] accumulator in workspace
// ---------------------------------------------------------------------------
__global__ void zero_agg(float* __restrict__ agg, long total) {
  long i = (long)blockIdx.x * blockDim.x + threadIdx.x;
  long stride = (long)gridDim.x * blockDim.x;
  for (; i < total; i += stride) agg[i] = 0.0f;
}

// ---------------------------------------------------------------------------
// Kernel 2: edge MLP via WMMA (16 edges per wave) + scatter atomics into agg.
//
// A matrix (16x32 f16, M=edge, K=input): lanes 0-15 hold row M=lane,
//   halves 0-7 = K 0-7 (src feats), halves 8-15 = K 16-23 (edge weight + pad);
//   lanes 16-31 hold row M=lane-16, halves 0-7 = K 8-15 (tgt feats),
//   halves 8-15 = K 24-31 (zero pad).   [per CDNA5 ISA 7.12.2]
// B matrix (32x16 f16, K x N=hidden): lanes 0-15 hold K 0-7,16-23 (half j<8: K=j, else K=j+8),
//   N = lane; lanes 16-31 hold K 8-15,24-31, N = lane-16.
// C/D (16x16 f32): lane group low -> M = vgpr, lanes 16-31 -> M = vgpr+8,
//   N = lane & 15.
// ---------------------------------------------------------------------------
__global__ __launch_bounds__(256) void edge_mlp_scatter(
    const float* __restrict__ nf,   // [B, N, 8]
    const float* __restrict__ ew,   // [B, E]
    const int*   __restrict__ ei,   // [2, E]
    const float* __restrict__ W1, const float* __restrict__ b1,   // 17x16, 16
    const float* __restrict__ W2, const float* __restrict__ b2,   // 16x16, 16
    const float* __restrict__ W3, const float* __restrict__ b3,   // 16x16, 16
    float* __restrict__ agg)        // [B, N, 16]
{
  __shared__ float lds_mat[WAVES_PER_BLOCK][256];  // 16x16 f32 transpose tile
  __shared__ int   lds_idx[WAVES_PER_BLOCK][32];   // src[0..15], tgt[0..15]

  const int  lane = threadIdx.x & 31;
  const int  wave = threadIdx.x >> 5;
  const int  n    = lane & 15;       // hidden-unit / column index for this lane
  const bool hiK  = lane >= 16;

  // ---- build B matrices (weights) in WMMA layout, once per wave ----
  v16h bw1, bw2, bw3;
#pragma unroll
  for (int j = 0; j < 16; ++j) {
    int k = (j < 8) ? ((hiK ? 8 : 0) + j) : (16 + (hiK ? 8 : 0) + (j - 8));
    float w1v = 0.0f;
    if (k < 17) w1v = W1[k * 16 + n];            // row 16 = edge-weight input
    bw1[j] = (_Float16)w1v;
    bw2[j] = (k >= 16) ? (_Float16)0.0f : (_Float16)W2[k * 16 + n];
    bw3[j] = (k >= 16) ? (_Float16)0.0f : (_Float16)W3[k * 16 + n];
  }
  const float bias1 = b1[n];
  const float bias2 = b2[n];
  const float bias3 = b3[n];

  const int waveGlobal = blockIdx.x * WAVES_PER_BLOCK + wave;
  const int waveStride = gridDim.x * WAVES_PER_BLOCK;

  for (int tile = waveGlobal; tile < NTILES; tile += waveStride) {
    const int bb = tile / TILES_PER_BATCH;
    const int t  = tile - bb * TILES_PER_BATCH;
    const int e0 = t * 16;

    // ---- gather: lane m<16 loads src endpoint, lane m+16 loads tgt ----
    const int  m  = lane & 15;
    const long e  = (long)e0 + m;                 // this lane's edge
    const long ec = e < E_ ? e : (long)E_ - 1;
    const int  sidx = ei[ec];
    const int  tidx = ei[(long)E_ + ec];
    if (lane < 16) {
      lds_idx[wave][m]      = sidx;
      lds_idx[wave][16 + m] = tidx;
    }
    const int gidx = hiK ? tidx : sidx;
    const float* fp = nf + ((size_t)bb * N_ + (size_t)gidx) * F_;
    const float4 f0 = *(const float4*)fp;
    const float4 f1 = *(const float4*)(fp + 4);
    const float  wgt = ew[(size_t)bb * E_ + ec];

    v16h a1;
    a1[0] = (_Float16)f0.x;  a1[1] = (_Float16)f0.y;
    a1[2] = (_Float16)f0.z;  a1[3] = (_Float16)f0.w;
    a1[4] = (_Float16)f1.x;  a1[5] = (_Float16)f1.y;
    a1[6] = (_Float16)f1.z;  a1[7] = (_Float16)f1.w;
    a1[8] = hiK ? (_Float16)0.0f : (_Float16)wgt;  // K=16 (edge weight)
#pragma unroll
    for (int j = 9; j < 16; ++j) a1[j] = (_Float16)0.0f;

    // ---- layer 1: h1 = relu(A1 x W1 + b1) ----
    v8f c = {};
    c = __builtin_amdgcn_wmma_f32_16x16x32_f16(false, a1, false, bw1,
                                               (short)0, c, false, false);
#pragma unroll
    for (int r = 0; r < 8; ++r) {
      float x = c[r] + bias1;
      c[r] = x > 0.0f ? x : 0.0f;
    }

    // ---- C-layout -> A-layout transpose through LDS ----
    const int mbase = hiK ? 8 : 0;
    float* tp = &lds_mat[wave][0];
#pragma unroll
    for (int r = 0; r < 8; ++r) tp[(mbase + r) * 16 + n] = c[r];
    asm volatile("s_wait_dscnt 0" ::: "memory");

    const int koff = hiK ? 8 : 0;
    v16h a2;
#pragma unroll
    for (int j = 0; j < 8; ++j) a2[j] = (_Float16)tp[m * 16 + koff + j];
#pragma unroll
    for (int j = 8; j < 16; ++j) a2[j] = (_Float16)0.0f;

    // ---- layer 2: h2 = relu(A2 x W2 + b2) ----
    v8f c2 = {};
    c2 = __builtin_amdgcn_wmma_f32_16x16x32_f16(false, a2, false, bw2,
                                                (short)0, c2, false, false);
#pragma unroll
    for (int r = 0; r < 8; ++r) {
      float x = c2[r] + bias2;
      c2[r] = x > 0.0f ? x : 0.0f;
    }

#pragma unroll
    for (int r = 0; r < 8; ++r) tp[(mbase + r) * 16 + n] = c2[r];
    asm volatile("s_wait_dscnt 0" ::: "memory");

    v16h a3;
#pragma unroll
    for (int j = 0; j < 8; ++j) a3[j] = (_Float16)tp[m * 16 + koff + j];
#pragma unroll
    for (int j = 8; j < 16; ++j) a3[j] = (_Float16)0.0f;

    // ---- layer 3: h = sigmoid(A3 x W3 + b3) ----
    v8f c3 = {};
    c3 = __builtin_amdgcn_wmma_f32_16x16x32_f16(false, a3, false, bw3,
                                                (short)0, c3, false, false);
#pragma unroll
    for (int r = 0; r < 8; ++r) c3[r] = fast_sigmoid(c3[r] + bias3);

    // ---- scatter: agg[tgt] += h ; agg[src] -= h (coalesced f32 atomics) ----
    const size_t aggBase = (size_t)bb * N_ * 16;
#pragma unroll
    for (int r = 0; r < 8; ++r) {
      const int  mm = mbase + r;
      const long ee = (long)e0 + mm;
      if (ee < E_) {
        const int sI = lds_idx[wave][mm];
        const int tI = lds_idx[wave][16 + mm];
        const float v = c3[r];
        atomicAdd(&agg[aggBase + (size_t)tI * 16 + n],  v);
        atomicAdd(&agg[aggBase + (size_t)sI * 16 + n], -v);
      }
    }
  }
}

// ---------------------------------------------------------------------------
// Kernel 3: node MLP  o = sigmoid(relu(agg @ Wn1 + bn1) @ Wn2 + bn2)
// one thread per (b, node); weight addresses are uniform -> scalar loads
// ---------------------------------------------------------------------------
__global__ void node_mlp(const float* __restrict__ agg,
                         const float* __restrict__ Wn1, const float* __restrict__ bn1,
                         const float* __restrict__ Wn2, const float* __restrict__ bn2,
                         float* __restrict__ out, long total)
{
  long i = (long)blockIdx.x * blockDim.x + threadIdx.x;
  if (i >= total) return;
  const float* row = agg + i * 16;
  float x[16];
#pragma unroll
  for (int k = 0; k < 16; ++k) x[k] = row[k];
  float acc = bn2[0];
#pragma unroll
  for (int j = 0; j < 12; ++j) {
    float h = bn1[j];
#pragma unroll
    for (int k = 0; k < 16; ++k) h = fmaf(x[k], Wn1[k * 12 + j], h);
    h = h > 0.0f ? h : 0.0f;
    acc = fmaf(h, Wn2[j], acc);
  }
  out[i] = fast_sigmoid(acc);
}

// ---------------------------------------------------------------------------
extern "C" void kernel_launch(void* const* d_in, const int* in_sizes, int n_in,
                              void* d_out, int out_size, void* d_ws, size_t ws_size,
                              hipStream_t stream) {
  const float* nf  = (const float*)d_in[0];   // node_features [B,N,8]
  const float* ew  = (const float*)d_in[1];   // edge_weight   [B,E]
  const int*   ei  = (const int*)  d_in[2];   // edge_index    [2,E]
  const float* W1  = (const float*)d_in[3];
  const float* b1  = (const float*)d_in[4];
  const float* W2  = (const float*)d_in[5];
  const float* b2  = (const float*)d_in[6];
  const float* W3  = (const float*)d_in[7];
  const float* b3  = (const float*)d_in[8];
  const float* Wn1 = (const float*)d_in[9];
  const float* bn1 = (const float*)d_in[10];
  const float* Wn2 = (const float*)d_in[11];
  const float* bn2 = (const float*)d_in[12];

  float* agg = (float*)d_ws;                  // [B, N, 16] = 12.8 MB
  const long aggElems = (long)B_ * N_ * 16;

  zero_agg<<<1024, 256, 0, stream>>>(agg, aggElems);

  edge_mlp_scatter<<<EDGE_BLOCKS, WAVES_PER_BLOCK * 32, 0, stream>>>(
      nf, ew, ei, W1, b1, W2, b2, W3, b3, agg);

  const long totalNodes = (long)B_ * N_;
  node_mlp<<<(int)((totalNodes + 255) / 256), 256, 0, stream>>>(
      agg, Wn1, bn1, Wn2, bn2, (float*)d_out, totalNodes);
}
